// BuildGFGRUStack_36283883716692
// MI455X (gfx1250) — matrix operand order repaired and checked by reading, not compiled, not run on hardware
//
#include <hip/hip_runtime.h>
#include <math.h>

// Gated-Feedback GRU stack for MI455X (gfx1250).
// Heavy path = 9x [B,256]x[256,256] GEMMs/layer fused into one WMMA kernel
// (V_WMMA_F32_16X16X32_BF16, fp32 accum), 7 accumulators per 16x16 tile.
// Weights (B operands) are staged in LDS with GLOBAL_LOAD_ASYNC_TO_LDS_B128,
// double-buffered per 32-wide K step and synchronized with s_wait_asynccnt.

#define NL 4
#define RNN 256
#define BATCH 16384
#define BUF_BYTES 36864   // 9 slices * 64 cols * 64B per K-step buffer

typedef __attribute__((ext_vector_type(16))) __bf16 v16bf;
typedef __attribute__((ext_vector_type(8)))  __bf16 v8bf;
typedef __attribute__((ext_vector_type(4)))  __bf16 v4bf;
typedef __attribute__((ext_vector_type(8)))  float  v8f;

// A (16x32, MxK): lane-half h holds K=16h..16h+15 contiguously in regs0-7.
static __device__ __forceinline__ v16bf load_a_frag(const __bf16* __restrict__ row,
                                                    int kb, int h) {
  v8bf lo = *(const v8bf*)(row + kb + 16 * h);
  v8bf hi = *(const v8bf*)(row + kb + 16 * h + 8);
  v16bf r;
#pragma unroll
  for (int j = 0; j < 8; ++j) { r[j] = lo[j]; r[j + 8] = hi[j]; }
  return r;
}

// ---- f32 -> bf16 bulk convert (4 elems/thread) ----
__global__ __launch_bounds__(256) void cvt_f32_bf16(const float* __restrict__ src,
                                                    __bf16* __restrict__ dst, int n4) {
  int i = blockIdx.x * blockDim.x + threadIdx.x;
  if (i < n4) {
    float4 v = ((const float4*)src)[i];
    v4bf o = {(__bf16)v.x, (__bf16)v.y, (__bf16)v.z, (__bf16)v.w};
    ((v4bf*)dst)[i] = o;
  }
}

// ---- per-layer global gates: g[b,s] = tanh(x_l . w_g[l,s] + h_stacked . w_ug[l,s] + b) ----
__global__ __launch_bounds__(256) void gates_kernel(
    const float* __restrict__ xl, const float* __restrict__ prevhs,
    const float* __restrict__ wg,  const float* __restrict__ wug,
    const float* __restrict__ bg,  const float* __restrict__ bug,
    float* __restrict__ gout) {
  const int wave = threadIdx.x >> 5;
  const int lane = threadIdx.x & 31;
  const int b = blockIdx.x * 8 + wave;
  float a0 = 0.f, a1 = 0.f, a2 = 0.f, a3 = 0.f;
  const float* xr = xl + (size_t)b * RNN;
#pragma unroll 2
  for (int k = lane; k < RNN; k += 32) {
    float xv = xr[k];
    a0 += xv * wg[k];           a1 += xv * wg[RNN + k];
    a2 += xv * wg[2 * RNN + k]; a3 += xv * wg[3 * RNN + k];
  }
#pragma unroll 2
  for (int k = lane; k < NL * RNN; k += 32) {
    int g2 = k >> 8;
    int r  = k & (RNN - 1);
    float hv = prevhs[((size_t)g2 * BATCH + b) * RNN + r];
    a0 += hv * wug[k];        a1 += hv * wug[1024 + k];
    a2 += hv * wug[2048 + k]; a3 += hv * wug[3072 + k];
  }
#pragma unroll
  for (int off = 16; off >= 1; off >>= 1) {
    a0 += __shfl_xor(a0, off, 32);
    a1 += __shfl_xor(a1, off, 32);
    a2 += __shfl_xor(a2, off, 32);
    a3 += __shfl_xor(a3, off, 32);
  }
  if (lane == 0) {
    float4 o;
    o.x = tanhf(a0 + bg[0] + bug[0]);
    o.y = tanhf(a1 + bg[1] + bug[1]);
    o.z = tanhf(a2 + bg[2] + bug[2]);
    o.w = tanhf(a3 + bg[3] + bug[3]);
    ((float4*)gout)[b] = o;
  }
}

// ---- main fused layer kernel ----
// Block: 256 threads = 8 waves; block tile 128 rows x 64 cols;
// wave owns 16 rows x 64 cols = 4 16x16 tiles, 7 fp32 accumulators each.
// B operands double-buffered in LDS via async copies.
__global__ __launch_bounds__(256) void gfgru_layer_kernel(
    const __bf16* __restrict__ xbf,    // [B,256]  bf16 layer input
    const __bf16* __restrict__ hbf,    // [L,B,256] bf16 prev_hs
    const __bf16* __restrict__ hbf_l,  // hbf + l*B*256
    const __bf16* __restrict__ w_i2h,  // [512,256] layer slice, bf16
    const __bf16* __restrict__ w_h2h,  // [512,256]
    const __bf16* __restrict__ w_j1j,  // [256,256]
    const __bf16* __restrict__ w_uij,  // [4,256,256]
    const float*  __restrict__ gbuf,   // [B,4]
    const float*  __restrict__ ph,     // prev_hs[l] fp32 [B,256]
    const float*  __restrict__ bi2h, const float* __restrict__ bh2h,  // [512]
    const float*  __restrict__ bj1j,                                  // [256]
    const float*  __restrict__ buij,                                  // [4,256]
    float*  __restrict__ out,          // [B,256] fp32 (d_out + l*B*R)
    __bf16* __restrict__ outbf) {      // [B,256] bf16 (next layer's x)
  __shared__ __bf16 smem[2 * BUF_BYTES / 2];  // 2 x 18432 bf16 (72KB)

  const int tid  = threadIdx.x;
  const int wave = tid >> 5;
  const int lane = tid & 31;
  const int h    = lane >> 4;
  const int nl   = lane & 15;
  const int m0   = blockIdx.x * 128 + wave * 16;
  const int n0   = blockIdx.y * 64;
  const int arow = m0 + nl;  // A-matrix row handled by this lane

  // ---- async B staging setup: thread tid covers (col = tid>>2, 16B part = tid&3)
  // LDS layout per buffer: slice s (9) x col (64) x 64B of K data.
  const int colg = n0 + (tid >> 2);
  const int part = (tid & 3) * 8;  // bf16 elements (16B)
  const __bf16* gsrc[9];
  gsrc[0] = w_i2h + (size_t)colg * RNN + part;           // z: Wz (x)
  gsrc[1] = w_i2h + (size_t)(256 + colg) * RNN + part;   // r: Wr (x)
  gsrc[2] = w_j1j + (size_t)colg * RNN + part;           // cand: Wc (x)
  gsrc[3] = w_h2h + (size_t)colg * RNN + part;           // z: Uz (h_l)
  gsrc[4] = w_h2h + (size_t)(256 + colg) * RNN + part;   // r: Ur (h_l)
#pragma unroll
  for (int g2 = 0; g2 < 4; ++g2)
    gsrc[5 + g2] = w_uij + ((size_t)g2 * RNN + colg) * RNN + part;

  const unsigned lds_base = (unsigned)(uintptr_t)(&smem[0]) + (unsigned)(tid * 16);

  auto fill = [&](int ksrc, int buf) {
#pragma unroll
    for (int s = 0; s < 9; ++s) {
      unsigned laddr = lds_base + (unsigned)(buf * BUF_BYTES + s * 4096);
      unsigned long long ga =
          (unsigned long long)(uintptr_t)gsrc[s] + (unsigned long long)(ksrc * 64);
      asm volatile("global_load_async_to_lds_b128 %0, %1, off"
                   :: "v"(laddr), "v"(ga) : "memory");
    }
  };

  v8f acc[7][4];
#pragma unroll
  for (int a = 0; a < 7; ++a)
#pragma unroll
    for (int t = 0; t < 4; ++t)
#pragma unroll
      for (int j = 0; j < 8; ++j) acc[a][t][j] = 0.0f;

  const __bf16* xrow  = xbf   + (size_t)arow * RNN;
  const __bf16* hlrow = hbf_l + (size_t)arow * RNN;

  fill(0, 0);  // prime buffer 0

  for (int ks = 0; ks < 8; ++ks) {  // K = 8 * 32 = 256 per chunk
    const int kb  = ks * 32;
    const int buf = ks & 1;

    // own-wave async ops for this step complete; barrier covers all waves.
    asm volatile("s_wait_asynccnt 0x0" ::: "memory");
    __syncthreads();
    if (ks < 7) fill(ks + 1, buf ^ 1);  // background copy overlaps compute

    v16bf a_x  = load_a_frag(xrow, kb, h);
    v16bf a_hl = load_a_frag(hlrow, kb, h);
    v16bf a_h[4];
#pragma unroll
    for (int g2 = 0; g2 < 4; ++g2)
      a_h[g2] = load_a_frag(hbf + ((size_t)g2 * BATCH + arow) * RNN, kb, h);

    // B frag from LDS: slice s, column t*16+nl, K-half h.
    const __bf16* sbase = smem + (size_t)buf * (BUF_BYTES / 2) + (size_t)h * 16;
#pragma unroll
    for (int t = 0; t < 4; ++t) {
      const int coff = (t * 16 + nl) * 32;  // elements per column = 32
      v16bf bf;
      bf = *(const v16bf*)(sbase + 0 * 2048 + coff);
      acc[0][t] = __builtin_amdgcn_wmma_f32_16x16x32_bf16(false, a_x, false, bf, (short)0, acc[0][t], false, false);
      bf = *(const v16bf*)(sbase + 1 * 2048 + coff);
      acc[1][t] = __builtin_amdgcn_wmma_f32_16x16x32_bf16(false, a_x, false, bf, (short)0, acc[1][t], false, false);
      bf = *(const v16bf*)(sbase + 2 * 2048 + coff);
      acc[2][t] = __builtin_amdgcn_wmma_f32_16x16x32_bf16(false, a_x, false, bf, (short)0, acc[2][t], false, false);
      bf = *(const v16bf*)(sbase + 3 * 2048 + coff);
      acc[0][t] = __builtin_amdgcn_wmma_f32_16x16x32_bf16(false, a_hl, false, bf, (short)0, acc[0][t], false, false);
      bf = *(const v16bf*)(sbase + 4 * 2048 + coff);
      acc[1][t] = __builtin_amdgcn_wmma_f32_16x16x32_bf16(false, a_hl, false, bf, (short)0, acc[1][t], false, false);
#pragma unroll
      for (int g2 = 0; g2 < 4; ++g2) {
        bf = *(const v16bf*)(sbase + (5 + g2) * 2048 + coff);
        acc[3 + g2][t] = __builtin_amdgcn_wmma_f32_16x16x32_bf16(false, a_h[g2], false, bf, (short)0, acc[3 + g2][t], false, false);
      }
    }
  }

  // Epilogue: C/D layout -> element j of v8f is (m = m0 + j + 8h, n = nl).
#pragma unroll
  for (int t = 0; t < 4; ++t) {
    const int q = n0 + t * 16 + nl;
    const float bu0 = buij[q], bu1 = buij[RNN + q], bu2 = buij[2 * RNN + q], bu3 = buij[3 * RNN + q];
    const float bz = bi2h[q] + bh2h[q];
    const float br = bi2h[256 + q] + bh2h[256 + q];
    const float bc = bj1j[q];
#pragma unroll
    for (int j = 0; j < 8; ++j) {
      const int m = m0 + j + 8 * h;
      float4 gv = ((const float4*)gbuf)[m];
      float su = gv.x * (acc[3][t][j] + bu0) + gv.y * (acc[4][t][j] + bu1) +
                 gv.z * (acc[5][t][j] + bu2) + gv.w * (acc[6][t][j] + bu3);
      float zt = 1.f / (1.f + __expf(-(acc[0][t][j] + bz)));
      float rt = 1.f / (1.f + __expf(-(acc[1][t][j] + br)));
      float cand = tanhf(acc[2][t][j] + bc + rt * su);
      float phv = ph[(size_t)m * RNN + q];
      float ht = (1.f - zt) * phv + zt * cand;
      out[(size_t)m * RNN + q] = ht;
      outbf[(size_t)m * RNN + q] = (__bf16)ht;
    }
  }
}

extern "C" void kernel_launch(void* const* d_in, const int* in_sizes, int n_in,
                              void* d_out, int out_size, void* d_ws, size_t ws_size,
                              hipStream_t stream) {
  const float* x      = (const float*)d_in[0];
  const float* prevhs = (const float*)d_in[1];
  const float* w_i2h  = (const float*)d_in[2];
  const float* b_i2h  = (const float*)d_in[3];
  const float* w_h2h  = (const float*)d_in[4];
  const float* b_h2h  = (const float*)d_in[5];
  const float* w_j1j  = (const float*)d_in[6];
  const float* b_j1j  = (const float*)d_in[7];
  const float* w_g    = (const float*)d_in[8];
  const float* b_g    = (const float*)d_in[9];
  const float* w_ug   = (const float*)d_in[10];
  const float* b_ug   = (const float*)d_in[11];
  const float* w_uij  = (const float*)d_in[12];
  const float* b_uij  = (const float*)d_in[13];
  float* out = (float*)d_out;

  // ---- scratch layout (~55 MB), 256B aligned slices ----
  char* ws = (char*)d_ws;
  size_t off = 0;
  auto take = [&](size_t bytes) -> char* {
    char* p = ws + off;
    off = (off + bytes + 255) & ~(size_t)255;
    return p;
  };
  const size_t n_i2h = (size_t)NL * 512 * RNN;
  const size_t n_j1j = (size_t)NL * RNN * RNN;
  const size_t n_uij = (size_t)NL * NL * RNN * RNN;
  const size_t n_h   = (size_t)NL * BATCH * RNN;
  const size_t n_x   = (size_t)BATCH * RNN;

  __bf16* wbf_i2h = (__bf16*)take(n_i2h * 2);
  __bf16* wbf_h2h = (__bf16*)take(n_i2h * 2);
  __bf16* wbf_j1j = (__bf16*)take(n_j1j * 2);
  __bf16* wbf_uij = (__bf16*)take(n_uij * 2);
  __bf16* hbf     = (__bf16*)take(n_h * 2);
  __bf16* xbf0    = (__bf16*)take(n_x * 2);
  __bf16* xbf1    = (__bf16*)take(n_x * 2);
  float*  gbuf    = (float*)take((size_t)BATCH * 4 * 4);

  auto cvt = [&](const float* s, __bf16* d, size_t n) {
    int n4 = (int)(n / 4);
    cvt_f32_bf16<<<(n4 + 255) / 256, 256, 0, stream>>>(s, d, n4);
  };
  cvt(w_i2h, wbf_i2h, n_i2h);
  cvt(w_h2h, wbf_h2h, n_i2h);
  cvt(w_j1j, wbf_j1j, n_j1j);
  cvt(w_uij, wbf_uij, n_uij);
  cvt(prevhs, hbf, n_h);
  cvt(x, xbf0, n_x);

  for (int l = 0; l < NL; ++l) {
    const float*  xl      = (l == 0) ? x : (out + (size_t)(l - 1) * BATCH * RNN);
    const __bf16* xbf_in  = (l & 1) ? xbf1 : xbf0;
    __bf16*       xbf_out = (l & 1) ? xbf0 : xbf1;

    gates_kernel<<<BATCH / 8, 256, 0, stream>>>(
        xl, prevhs,
        w_g + (size_t)l * NL * RNN, w_ug + (size_t)l * NL * NL * RNN,
        b_g + (size_t)l * NL,       b_ug + (size_t)l * NL,
        gbuf);

    dim3 grid(BATCH / 128, RNN / 64);
    gfgru_layer_kernel<<<grid, 256, 0, stream>>>(
        xbf_in, hbf, hbf + (size_t)l * BATCH * RNN,
        wbf_i2h + (size_t)l * 512 * RNN,
        wbf_h2h + (size_t)l * 512 * RNN,
        wbf_j1j + (size_t)l * RNN * RNN,
        wbf_uij + (size_t)l * NL * RNN * RNN,
        gbuf,
        prevhs + (size_t)l * BATCH * RNN,
        b_i2h + (size_t)l * 512, b_h2h + (size_t)l * 512,
        b_j1j + (size_t)l * RNN, b_uij + (size_t)l * NL * RNN,
        out + (size_t)l * BATCH * RNN,
        xbf_out);
  }
  (void)in_sizes; (void)n_in; (void)out_size; (void)ws_size;
}
